// CausalAttention_15015205667430
// MI455X (gfx1250) — compile-verified, matrix operand order checked by reading, not run on hardware
//
#include <hip/hip_runtime.h>
#include <hip/hip_bf16.h>
#include <math.h>
#include <stdint.h>

typedef _Float16 half_t;
typedef __attribute__((ext_vector_type(16))) _Float16 v16h;
typedef __attribute__((ext_vector_type(8)))  _Float16 v8h;
typedef __attribute__((ext_vector_type(8)))  float    v8f;
typedef __attribute__((ext_vector_type(4)))  float    v4f;
typedef __attribute__((ext_vector_type(4)))  unsigned u32x4;
typedef __attribute__((ext_vector_type(8)))  int      i32x8;
typedef __attribute__((ext_vector_type(4)))  int      i32x4;

#define D_MODEL 1024
#define SEQ     4096
#define BATCH   4
#define NROWS   (BATCH * SEQ)   // 16384

// ---- CDNA5 async global->LDS copy (ASYNCcnt path), 16B per lane ----
// Flat shared-pointer low 32 bits == LDS allocation-relative offset
// (aperture base occupies addr[63:32]).
__device__ __forceinline__ void async_ld_b128(void* lds, const void* gptr) {
    unsigned l = (unsigned)(uintptr_t)lds;
    asm volatile("global_load_async_to_lds_b128 %0, %1, off"
                 :: "v"(l), "v"(gptr) : "memory");
}
#define ASYNC_WAIT(n)  asm volatile("s_wait_asynccnt " #n ::: "memory")
#define TENSOR_WAIT(n) asm volatile("s_wait_tensorcnt " #n ::: "memory")

// ---- CDNA5 Tensor Data Mover: one-descriptor 2D tile DMA (TENSORcnt) ----
#ifdef __has_builtin
#if defined(__HIP_DEVICE_COMPILE__) && __has_builtin(__builtin_amdgcn_tensor_load_to_lds)
#define HAVE_TDM 1
#endif
#endif
#ifndef HAVE_TDM
#define HAVE_TDM 0
#endif

#if HAVE_TDM
// 2D tile load: td0 elements per row, td1 rows, row stride stride0 (elements),
// 2-byte elements. LDS padding: after 2^(padi+1) DWORDs insert (pada+1) DWORDs.
__device__ __forceinline__ void tdm_load_2d(unsigned lds_byte, const void* g,
                                            unsigned td0, unsigned td1,
                                            unsigned stride0,
                                            unsigned padi, unsigned pada) {
    unsigned long long ga = (unsigned long long)(uintptr_t)g;
    u32x4 g0;
    g0[0] = 1u;                                   // count=1, user descriptor
    g0[1] = lds_byte;                             // lds_addr
    g0[2] = (unsigned)ga;                         // global_addr[31:0]
    g0[3] = (unsigned)(ga >> 32) | (2u << 30);    // global_addr[56:32] | type=2
    i32x8 g1;
    g1[0] = (int)((1u << 16)                      // data_size = 2 bytes
                  | (1u << 20)                    // pad_enable
                  | (padi << 22) | (pada << 25));
    g1[1] = (int)((td0 & 0xFFFFu) << 16);                       // tensor_dim0 lo
    g1[2] = (int)(((td0 >> 16) & 0xFFFFu) | ((td1 & 0xFFFFu) << 16)); // d0 hi|d1 lo
    g1[3] = (int)(((td1 >> 16) & 0xFFFFu) | ((td0 & 0xFFFFu) << 16)); // d1 hi|tile_dim0
    g1[4] = (int)(td1 & 0xFFFFu);                 // tile_dim1 (tile_dim2 = 0)
    g1[5] = (int)stride0;                         // tensor_dim0_stride[31:0]
    g1[6] = 0;
    g1[7] = 0;
    i32x4 gz = {0, 0, 0, 0};
#if __clang_major__ >= 23
    i32x8 z8 = {0, 0, 0, 0, 0, 0, 0, 0};
    __builtin_amdgcn_tensor_load_to_lds(g0, g1, gz, gz, z8, 0);
#else
    __builtin_amdgcn_tensor_load_to_lds(g0, g1, gz, gz, 0);
#endif
}
#endif

// Load a 16x32 f16 WMMA A/B fragment from LDS.
// Per ISA layout: lane (lm = lane&15, lh = lane>>4) holds, for vector elem e:
//   K = (e>>3)*16 + lh*8 + (e&7)
// => two contiguous 16B chunks at [off, off+8) and [off+16, off+24), off = lh*8.
__device__ __forceinline__ v16h frag_ld(const half_t* row, int off) {
    v8h lo = *(const v8h*)(row + off);
    v8h hi = *(const v8h*)(row + off + 16);
    v16h r;
#pragma unroll
    for (int i = 0; i < 8; i++) { r[i] = lo[i]; r[i + 8] = hi[i]; }
    return r;
}

__device__ __forceinline__ v8f wmma_f16(v16h a, v16h b, v8f c) {
    return __builtin_amdgcn_wmma_f32_16x16x32_f16(false, a, false, b, (short)0, c,
                                                  false, false);
}

// ---------------------------------------------------------------------------
// Kernel 1: QKV projection GEMM.  C[16384 x 1024] = x[16384 x 1024] * W
// grid.z selects the weight (0=Wq, 1=Wk, 2=Wv).  Q,K stored f16 row-major;
// V stored transposed per batch: vt[b][d][s].
// ---------------------------------------------------------------------------
__global__ __launch_bounds__(256)
void qkv_gemm_kernel(const float* __restrict__ x,
                     const float* __restrict__ Wq,
                     const float* __restrict__ Wk,
                     const float* __restrict__ Wv,
                     half_t* __restrict__ qg,
                     half_t* __restrict__ kg,
                     half_t* __restrict__ vtg) {
    __shared__ half_t As[64 * 40];   // 64 rows x 32 k (+8 pad)
    __shared__ half_t Bs[64 * 40];   // 64 n-rows x 32 k (+8 pad), W transposed

    const int z = blockIdx.z;
    const float* __restrict__ W = (z == 0) ? Wq : (z == 1) ? Wk : Wv;

    const int m0 = blockIdx.x * 64;
    const int n0 = blockIdx.y * 64;

    const int t    = threadIdx.x;
    const int lane = t & 31;
    const int wave = t >> 5;
    const int mi   = wave & 3;     // m-subtile 0..3
    const int ng   = wave >> 2;    // n-group 0..1 (each covers two 16-subtiles)
    const int lm   = lane & 15;
    const int lh   = lane >> 4;

    v8f acc0 = {};
    v8f acc1 = {};

    for (int k0 = 0; k0 < D_MODEL; k0 += 32) {
        __syncthreads();
        // stage x tile (fp32 -> f16), row-major
        {
            int row = t >> 2;
            int seg = (t & 3) * 8;
            const float* gx = x + (size_t)(m0 + row) * D_MODEL + k0 + seg;
            v4f f0 = *(const v4f*)gx;
            v4f f1 = *(const v4f*)(gx + 4);
            half_t* dst = &As[row * 40 + seg];
#pragma unroll
            for (int i = 0; i < 4; i++) {
                dst[i]     = (half_t)f0[i];
                dst[4 + i] = (half_t)f1[i];
            }
        }
        // stage W tile transposed: Bs[n][kk]
        {
            int kk = t >> 3;
            int nb = (t & 7) * 8;
            const float* gw = W + (size_t)(k0 + kk) * D_MODEL + n0 + nb;
            v4f w0 = *(const v4f*)gw;
            v4f w1 = *(const v4f*)(gw + 4);
#pragma unroll
            for (int i = 0; i < 4; i++) {
                Bs[(nb + i) * 40 + kk]     = (half_t)w0[i];
                Bs[(nb + 4 + i) * 40 + kk] = (half_t)w1[i];
            }
        }
        __syncthreads();

        v16h a  = frag_ld(&As[(mi * 16 + lm) * 40], lh * 8);
        v16h b0 = frag_ld(&Bs[(ng * 32 + lm) * 40], lh * 8);
        v16h b1 = frag_ld(&Bs[(ng * 32 + 16 + lm) * 40], lh * 8);
        acc0 = wmma_f16(a, b0, acc0);
        acc1 = wmma_f16(a, b1, acc1);
    }

    // epilogue: write f16
    const int col0 = n0 + ng * 32 + lm;
    const int col1 = col0 + 16;
#pragma unroll
    for (int r = 0; r < 8; r++) {
        int grow = m0 + mi * 16 + r + 8 * lh;
        if (z == 0) {
            qg[(size_t)grow * D_MODEL + col0] = (half_t)acc0[r];
            qg[(size_t)grow * D_MODEL + col1] = (half_t)acc1[r];
        } else if (z == 1) {
            kg[(size_t)grow * D_MODEL + col0] = (half_t)acc0[r];
            kg[(size_t)grow * D_MODEL + col1] = (half_t)acc1[r];
        } else {
            int b = grow >> 12;        // /4096
            int s = grow & 4095;
            vtg[((size_t)b * D_MODEL + col0) * SEQ + s] = (half_t)acc0[r];
            vtg[((size_t)b * D_MODEL + col1) * SEQ + s] = (half_t)acc1[r];
        }
    }
}

// ---------------------------------------------------------------------------
// Kernel 2: causal flash attention.  One block = 32 query rows of one batch.
// 512 threads = 16 waves: (mi = wave&1) x (dgrp = wave>>1 in 0..7).
// Each wave accumulates O[16 x 128] fp32 (two 64-col halves at vc*512+dgrp*64).
// K and V^T tiles stream via the Tensor Data Mover (double-buffered); falls
// back to per-lane async global->LDS copies if the TDM builtin is absent.
// ---------------------------------------------------------------------------
__global__ __launch_bounds__(512)
void flash_attn_kernel(const half_t* __restrict__ qg,
                       const half_t* __restrict__ kg,
                       const half_t* __restrict__ vtg,
                       float* __restrict__ out) {
    __shared__ half_t Qs[32 * 1032];      // 66 KB  (32 rows x 1024 d, +8 pad)
    __shared__ half_t Ks[2][32 * 264];    // 2x17 KB (32 keys x 256-d chunk)
    __shared__ half_t Vts[2][512 * 40];   // 2x40 KB (512 d-rows x 32 keys)
    __shared__ float  Ssum[32 * 32];      // score tile reduction
    __shared__ half_t Ps[32 * 40];        // P = exp(S - m), f16
    __shared__ float  rowM[32], rowL[32], rowScale[32];

    const int q0 = blockIdx.x * 32;
    const int b  = blockIdx.y;

    const int t    = threadIdx.x;
    const int lane = t & 31;
    const int wave = t >> 5;
    const int mi   = wave & 1;          // query-row subtile
    const int dgrp = wave >> 1;         // 0..7: 64-wide output d-slice
    const int niS  = (wave >> 1) & 1;   // S-phase key subtile
    const int ksS  = wave >> 2;         // S-phase 64-wide k-dim slice
    const int lm   = lane & 15;
    const int lh   = lane >> 4;

    if (t < 32) { rowM[t] = -1e30f; rowL[t] = 0.0f; }

    // async-load Q tile (persistent for the whole block): 8 x 16B per thread
    {
        int row  = t >> 4;
        int base = (t & 15) * 64;
        const half_t* gq = qg + ((size_t)(b * SEQ + q0 + row)) * D_MODEL + base;
        half_t* dst = &Qs[row * 1032 + base];
#pragma unroll
        for (int i = 0; i < 8; i++) async_ld_b128(dst + i * 8, gq + i * 8);
    }
    ASYNC_WAIT(0);   // Q resident (per-wave); published by first barrier below

    // per-thread async staging indices (fallback path)
    const int krow = t >> 4;
    const int kseg = (t & 15) * 16;

    v8f acc[2][4] = {};   // O accumulator: [vc half][16-col subtile]
    const float scale = 0.03125f;   // 1/sqrt(1024)

    const int ntiles = q0 / 32 + 1;
    for (int kt = 0; kt < ntiles; kt++) {
        const int k0 = kt * 32;

        // issue K chunk 0, then V chunk 0 (in-order completion)
#if HAVE_TDM
        if (wave == 0) {
            tdm_load_2d((unsigned)(uintptr_t)&Ks[0][0],
                        kg + ((size_t)(b * SEQ + k0)) * D_MODEL,
                        256, 32, D_MODEL, 6, 3);      // 512B rows -> +16B pad
            tdm_load_2d((unsigned)(uintptr_t)&Vts[0][0],
                        vtg + ((size_t)b * D_MODEL) * SEQ + k0,
                        32, 512, SEQ, 3, 3);          // 64B rows  -> +16B pad
        }
#else
        {
            const half_t* gk =
                kg + ((size_t)(b * SEQ + k0 + krow)) * D_MODEL + kseg;
            half_t* dst = &Ks[0][krow * 264 + kseg];
            async_ld_b128(dst, gk);
            async_ld_b128(dst + 8, gk + 8);
        }
        {
            const half_t* gv = vtg + ((size_t)b * D_MODEL + t) * SEQ + k0;
            half_t* dst = &Vts[0][t * 40];
#pragma unroll
            for (int i = 0; i < 4; i++) async_ld_b128(dst + i * 8, gv + i * 8);
        }
#endif
        // zero score reduction buffer (visible after first barrier below)
        for (int i = t; i < 1024; i += 512) Ssum[i] = 0.0f;

        // ---- S = Q K^T, inner dim 1024 in 256-wide chunks, double-buffered ----
        v8f sacc = {};
#pragma unroll
        for (int dc = 0; dc < 4; dc++) {
            // wait for K[dc]; newest in-flight ops allowed to remain:
            //   dc==0: V0   dc==3: V1   else: none   (issues happen post-barrier)
#if HAVE_TDM
            if (wave == 0) {
                if (dc == 0 || dc == 3) { TENSOR_WAIT(1); } else { TENSOR_WAIT(0); }
            }
#else
            if (dc == 0 || dc == 3) { ASYNC_WAIT(4); } else { ASYNC_WAIT(0); }
#endif
            __syncthreads();
            // prefetch next K chunk (and V chunk 1 near the end of the phase);
            // issuing after the barrier proves the target buffer's readers done
#if HAVE_TDM
            if (wave == 0) {
                if (dc < 3)
                    tdm_load_2d((unsigned)(uintptr_t)&Ks[(dc + 1) & 1][0],
                                kg + ((size_t)(b * SEQ + k0)) * D_MODEL +
                                    (dc + 1) * 256,
                                256, 32, D_MODEL, 6, 3);
                if (dc == 2)
                    tdm_load_2d((unsigned)(uintptr_t)&Vts[1][0],
                                vtg + ((size_t)b * D_MODEL + 512) * SEQ + k0,
                                32, 512, SEQ, 3, 3);
            }
#else
            if (dc < 3) {
                const half_t* gk = kg + ((size_t)(b * SEQ + k0 + krow)) * D_MODEL +
                                   (dc + 1) * 256 + kseg;
                half_t* dst = &Ks[(dc + 1) & 1][krow * 264 + kseg];
                async_ld_b128(dst, gk);
                async_ld_b128(dst + 8, gk + 8);
            }
            if (dc == 2) {
                const half_t* gv =
                    vtg + ((size_t)b * D_MODEL + 512 + t) * SEQ + k0;
                half_t* dst = &Vts[1][t * 40];
#pragma unroll
                for (int i = 0; i < 4; i++) async_ld_b128(dst + i * 8, gv + i * 8);
            }
#endif
            // compute on K[dc]
            const half_t* qrow = &Qs[(mi * 16 + lm) * 1032 + dc * 256 + ksS * 64];
            const half_t* krw  = &Ks[dc & 1][(niS * 16 + lm) * 264 + ksS * 64];
#pragma unroll
            for (int st = 0; st < 2; st++) {
                v16h a  = frag_ld(qrow + st * 32, lh * 8);
                v16h bb = frag_ld(krw + st * 32, lh * 8);
                sacc = wmma_f16(a, bb, sacc);
            }
        }
        // reduce partial S tiles across the 4 k-slice waves
        {
            int srow = mi * 16 + 8 * lh;
            int scol = niS * 16 + lm;
#pragma unroll
            for (int r = 0; r < 8; r++)
                atomicAdd(&Ssum[(srow + r) * 32 + scol], sacc[r]);
        }
        __syncthreads();

        // ---- online softmax (one thread per query row) ----
        if (t < 32) {
            int m      = t;
            int qidx   = q0 + m;
            int nvalid = qidx - k0 + 1;
            if (nvalid > 32) nvalid = 32;
            float mx = rowM[m];
            for (int j = 0; j < nvalid; j++) {
                float v = Ssum[m * 32 + j] * scale;
                mx = fmaxf(mx, v);
            }
            float f = __expf(rowM[m] - mx);
            float l = rowL[m] * f;
            for (int j = 0; j < 32; j++) {
                float p = (j < nvalid) ? __expf(Ssum[m * 32 + j] * scale - mx) : 0.0f;
                l += (j < nvalid) ? p : 0.0f;
                Ps[m * 40 + j] = (half_t)p;
            }
            rowM[m] = mx;
            rowL[m] = l;
            rowScale[m] = f;
        }
        __syncthreads();

        // rescale existing O accumulator
        float fr[8];
#pragma unroll
        for (int r = 0; r < 8; r++) fr[r] = rowScale[mi * 16 + 8 * lh + r];
#pragma unroll
        for (int c = 0; c < 2; c++)
#pragma unroll
            for (int j = 0; j < 4; j++)
#pragma unroll
                for (int r = 0; r < 8; r++) acc[c][j][r] *= fr[r];

        // ---- O += P * V: chunk 0 (already resident), then chunk 1 ----
#pragma unroll
        for (int vc = 0; vc < 2; vc++) {
#if HAVE_TDM
            if (vc == 1 && wave == 0) { TENSOR_WAIT(0); }
#else
            if (vc == 1) { ASYNC_WAIT(0); }
#endif
            __syncthreads();                  // all waves' V writes + Ps ready
            v16h a = frag_ld(&Ps[(mi * 16 + lm) * 40], lh * 8);
#pragma unroll
            for (int j = 0; j < 4; j++) {
                const half_t* vrow = &Vts[vc][(dgrp * 64 + j * 16 + lm) * 40];
                v16h bb = frag_ld(vrow, lh * 8);
                acc[vc][j] = wmma_f16(a, bb, acc[vc][j]);
            }
        }
    }

    // ---- epilogue: divide by softmax denominator, store fp32 ----
    __syncthreads();
    float invl[8];
#pragma unroll
    for (int r = 0; r < 8; r++) invl[r] = 1.0f / rowL[mi * 16 + 8 * lh + r];
#pragma unroll
    for (int c = 0; c < 2; c++)
#pragma unroll
        for (int j = 0; j < 4; j++)
#pragma unroll
            for (int r = 0; r < 8; r++) {
                int row = q0 + mi * 16 + r + 8 * lh;
                int col = c * 512 + dgrp * 64 + j * 16 + lm;
                out[((size_t)b * SEQ + row) * D_MODEL + col] = acc[c][j][r] * invl[r];
            }
}

extern "C" void kernel_launch(void* const* d_in, const int* in_sizes, int n_in,
                              void* d_out, int out_size, void* d_ws, size_t ws_size,
                              hipStream_t stream) {
    const float* x  = (const float*)d_in[0];
    const float* Wq = (const float*)d_in[1];
    const float* Wk = (const float*)d_in[2];
    const float* Wv = (const float*)d_in[3];
    float* out = (float*)d_out;

    // workspace: Q, K (row-major f16) and V^T ([b][d][s] f16): 3 x 32 MB
    half_t* qg  = (half_t*)d_ws;
    half_t* kg  = qg + (size_t)NROWS * D_MODEL;
    half_t* vtg = kg + (size_t)NROWS * D_MODEL;

    dim3 g1(NROWS / 64, D_MODEL / 64, 3);
    qkv_gemm_kernel<<<g1, 256, 0, stream>>>(x, Wq, Wk, Wv, qg, kg, vtg);

    dim3 g2(SEQ / 32, BATCH, 1);
    flash_attn_kernel<<<g2, 512, 0, stream>>>(qg, kg, vtg, out);
}
